// Mul_Attention_65575560675598
// MI455X (gfx1250) — matrix operand order repaired and checked by reading, not compiled, not run on hardware
//
#include <hip/hip_runtime.h>

typedef _Float16 half8 __attribute__((ext_vector_type(8)));
typedef _Float16 v16h  __attribute__((ext_vector_type(16)));
typedef float    v8f   __attribute__((ext_vector_type(8)));

#define LSEQ 150
#define LP   160          // padded to multiple of 16 (M/N) and 32 (K)
#define KCH  5            // 5 K-chunks of 32 -> K=160 (zero padded)
#define NTL  10           // 10 N tiles of 16
#define NTHREADS 320      // 10 wave32

// LDS layout (bytes):
//   S   : 160*160 f32  = 102400
//   Ah  : 160*160 f16  =  51200
//   Bh  : 160*160 f16  =  51200
//   10 f32[160] arrays =   6400
//   nums[2] f32        =      8
#define SMEM_BYTES (102400 + 51200 + 51200 + 10*LP*4 + 8)

__global__ __launch_bounds__(NTHREADS)
void mul_attention_fused(const float* __restrict__ x0,
                         const float* __restrict__ x1,
                         const int*   __restrict__ mask0,
                         const int*   __restrict__ mask1,
                         float*       __restrict__ out,
                         int nbatch)
{
    extern __shared__ char smem[];
    float*    S      = (float*)smem;                      // [160][160]
    _Float16* Ah     = (_Float16*)(smem + 102400);        // masked0 f16 [160][160]
    _Float16* Bh     = (_Float16*)(smem + 153600);        // masked1 f16 [160][160]
    float*    rowmax = (float*)(smem + 204800);
    float*    rowinv = rowmax + LP;
    float*    colmax = rowinv + LP;
    float*    colinv = colmax + LP;
    float*    alphaL = colinv + LP;
    float*    betaL  = alphaL + LP;
    float*    gAL    = betaL  + LP;
    float*    gBL    = gAL    + LP;
    float*    m0f    = gBL    + LP;
    float*    m1f    = m0f    + LP;
    float*    nums   = m1f    + LP;   // [2]

    const int tid = threadIdx.x;
    const int b   = blockIdx.x;
    const size_t xbase = (size_t)b * (LSEQ * LSEQ);

    // ---- masks to LDS (zero-padded) ----
    if (tid < LP) {
        float v0 = 0.f, v1 = 0.f;
        if (tid < LSEQ) {
            v0 = (float)mask0[b * LSEQ + tid];
            v1 = (float)mask1[b * LSEQ + tid];
        }
        m0f[tid] = v0;
        m1f[tid] = v1;
    }
    __syncthreads();
    if (tid == 0) { float s = 0.f; for (int i = 0; i < LSEQ; ++i) s += m0f[i]; nums[0] = s; }
    if (tid == 1) { float s = 0.f; for (int i = 0; i < LSEQ; ++i) s += m1f[i]; nums[1] = s; }

    // ---- masked operands -> f16 LDS, zero padded to 160x160 ----
    for (int idx = tid; idx < LP * LP; idx += NTHREADS) {
        int r = idx / LP;
        int d = idx - r * LP;
        float a = 0.f, c = 0.f;
        if (r < LSEQ && d < LSEQ) {
            a = x0[xbase + r * LSEQ + d] * m0f[r];
            c = x1[xbase + r * LSEQ + d] * m1f[r];
        }
        Ah[idx] = (_Float16)a;
        Bh[idx] = (_Float16)c;
    }
    __syncthreads();

    // ---- WMMA GEMM: S[i,j] = sum_d Ah[i,d] * Bh[j,d] ----
    {
        const int wave = tid >> 5;
        const int lane = tid & 31;
        const int hl   = lane >> 4;   // half-wave select
        const int l15  = lane & 15;

        // A fragments (16x32 f16 per chunk): lane l15 = row, half-wave splits K
        union { v16h v; half8 h[2]; } af[KCH];
        const int arow = wave * 16 + l15;
        #pragma unroll
        for (int kc = 0; kc < KCH; ++kc) {
            const _Float16* p = Ah + arow * LP + kc * 32 + hl * 16;
            af[kc].h[0] = *(const half8*)(p);        // K = base+0..7   (or +16..23)
            af[kc].h[1] = *(const half8*)(p + 8);    // K = base+8..15  (or +24..31)
        }

        for (int nt = 0; nt < NTL; ++nt) {
            v8f acc = {};
            const int bcol = nt * 16 + l15;          // N index
            #pragma unroll
            for (int kc = 0; kc < KCH; ++kc) {
                // B (32x16): lane = column, 16 contiguous K per half-wave
                v16h bf = *(const v16h*)(Bh + bcol * LP + kc * 32 + hl * 16);
                acc = __builtin_amdgcn_wmma_f32_16x16x32_f16(
                          false, af[kc].v, false, bf, (short)0, acc, false, false);
            }
            // C layout: lane 0-15 N=lane, VGPR r -> M=r (lo half) / r+8 (hi half)
            const int srow = wave * 16 + hl * 8;
            const int scol = nt * 16 + l15;
            #pragma unroll
            for (int r = 0; r < 8; ++r)
                S[(srow + r) * LP + scol] = acc[r];
        }
    }
    __syncthreads();

    // ---- softmax statistics (rows for B_weight, cols for A_weight) ----
    if (tid < LSEQ) {                       // row i
        const int i = tid;
        float mx = -3.4e38f;
        for (int j = 0; j < LSEQ; ++j) mx = fmaxf(mx, S[i * LP + j]);
        float sm = 0.f;
        for (int j = 0; j < LSEQ; ++j) sm += __expf(S[i * LP + j] - mx);
        rowmax[i] = mx;
        rowinv[i] = 1.f / sm;
    } else if (tid >= LP && tid < LP + LSEQ) {  // col j
        const int j = tid - LP;
        float mx = -3.4e38f;
        for (int i = 0; i < LSEQ; ++i) mx = fmaxf(mx, S[i * LP + j]);
        float sm = 0.f;
        for (int i = 0; i < LSEQ; ++i) sm += __expf(S[i * LP + j] - mx);
        colmax[j] = mx;
        colinv[j] = 1.f / sm;
    }
    __syncthreads();

    // ---- alpha[i] = mean_j A_weight[i,j]; beta[j] = mean_i B_weight[i,j] ----
    if (tid < LSEQ) {
        const int i = tid;
        float s = 0.f;
        for (int j = 0; j < LSEQ; ++j)
            s += __expf(S[i * LP + j] - colmax[j]) * colinv[j];
        alphaL[i] = s * (1.0f / LSEQ);
    } else if (tid >= LP && tid < LP + LSEQ) {
        const int j = tid - LP;
        float s = 0.f;
        for (int i = 0; i < LSEQ; ++i)
            s += __expf(S[i * LP + j] - rowmax[i]) * rowinv[i];
        betaL[j] = s * (1.0f / LSEQ);
    }
    __syncthreads();

    // ---- gamma_A[k] = sum_j beta[j]*Bw[k,j]; gamma_B[k] = sum_j alpha[j]*Aw[j,k] ----
    if (tid < LSEQ) {
        const int k = tid;
        const float rm = rowmax[k];
        float s = 0.f;
        for (int j = 0; j < LSEQ; ++j)
            s += betaL[j] * __expf(S[k * LP + j] - rm);
        gAL[k] = s * rowinv[k];
    } else if (tid >= LP && tid < LP + LSEQ) {
        const int k = tid - LP;
        const float cm = colmax[k];
        float s = 0.f;
        for (int j = 0; j < LSEQ; ++j)
            s += alphaL[j] * __expf(S[j * LP + k] - cm);
        gBL[k] = s * colinv[k];
    }
    __syncthreads();

    // ---- outputs: A = gamma_A @ x0 / nums0 ; Bv = gamma_B @ x1 / nums1 ----
    const size_t BL = (size_t)nbatch * LSEQ;
    if (tid < LSEQ) {
        const int d = tid;
        float s = 0.f;
        for (int k = 0; k < LSEQ; ++k)
            s += gAL[k] * x0[xbase + k * LSEQ + d];     // coalesced across d
        out[(size_t)b * LSEQ + d]          = s / nums[0];
        out[2 * BL + (size_t)b * LSEQ + d] = alphaL[d];
    } else if (tid >= LP && tid < LP + LSEQ) {
        const int d = tid - LP;
        float s = 0.f;
        for (int k = 0; k < LSEQ; ++k)
            s += gBL[k] * x1[xbase + k * LSEQ + d];
        out[BL + (size_t)b * LSEQ + d]     = s / nums[1];
        out[3 * BL + (size_t)b * LSEQ + d] = betaL[d];
    }
}

extern "C" void kernel_launch(void* const* d_in, const int* in_sizes, int n_in,
                              void* d_out, int out_size, void* d_ws, size_t ws_size,
                              hipStream_t stream) {
    (void)n_in; (void)out_size; (void)d_ws; (void)ws_size;
    const float* x0 = (const float*)d_in[0];
    const float* x1 = (const float*)d_in[1];
    const int*   m0 = (const int*)d_in[2];
    const int*   m1 = (const int*)d_in[3];
    float* out = (float*)d_out;

    const int nbatch = in_sizes[2] / LSEQ;   // mask0 is [B, L]

    // Allow >64KB dynamic LDS (CDNA5 WGP has 320KB). Deterministic, idempotent.
    (void)hipFuncSetAttribute(reinterpret_cast<const void*>(mul_attention_fused),
                              hipFuncAttributeMaxDynamicSharedMemorySize,
                              (int)SMEM_BYTES);

    mul_attention_fused<<<nbatch, NTHREADS, SMEM_BYTES, stream>>>(
        x0, x1, m0, m1, out, nbatch);
}
